// VQ_Sinkhorn_Knopp_21431886807705
// MI455X (gfx1250) — matrix operand order repaired and checked by reading, not compiled, not run on hardware
//
#include <hip/hip_runtime.h>
#include <cmath>
#include <cstdint>

typedef __attribute__((ext_vector_type(16))) _Float16 v16h;
typedef __attribute__((ext_vector_type(8)))  float    v8f;

#define INV_EPS 20.0f  /* 1 / SK_EPS (0.05) */

static __device__ __forceinline__ float wave_sum(float v) {
#pragma unroll
  for (int m = 16; m >= 1; m >>= 1) v += __shfl_xor(v, m, 32);
  return v;
}

// ---------------- row L2 normalize: fp32 (optional) + f16 copies ----------------
__global__ void k_l2norm(const float* __restrict__ src, float* __restrict__ dst32,
                         _Float16* __restrict__ dst16, int rows, int D) {
  int wave = threadIdx.x >> 5, lane = threadIdx.x & 31;
  int row = blockIdx.x * (blockDim.x >> 5) + wave;
  if (row >= rows) return;
  const float* s = src + (size_t)row * D;
  float ss = 0.f;
  for (int i = lane; i < D; i += 32) { float v = s[i]; ss += v * v; }
  ss = wave_sum(ss);
  float scale = 1.0f / fmaxf(sqrtf(ss), 1e-12f);
  for (int i = lane; i < D; i += 32) {
    float v = s[i] * scale;
    if (dst32) dst32[(size_t)row * D + i] = v;
    dst16[(size_t)row * D + i] = (_Float16)v;
  }
}

// ---------------- score = (x_hat . keys_hat^T) * invKT via WMMA f16 ----------------
union Frag { v16h h; uint4 u4[2]; };

__global__ void k_score_wmma(const _Float16* __restrict__ A,  // [M][D]
                             const _Float16* __restrict__ B,  // [N][D]
                             float* __restrict__ score,       // [M][N]
                             int M, int N, int D, float invKT) {
  const int wave = threadIdx.x >> 5;
  const int lane = threadIdx.x & 31;
  const int r  = lane & 15;
  const int hi = lane >> 4;
  const int mBase = (blockIdx.y * (blockDim.x >> 5) + wave) * 32;
  const int nBase = blockIdx.x * 32;

  const uint4* A0 = (const uint4*)(A + (size_t)(mBase +      r) * D);
  const uint4* A1 = (const uint4*)(A + (size_t)(mBase + 16 + r) * D);
  const uint4* B0 = (const uint4*)(B + (size_t)(nBase +      r) * D);
  const uint4* B1 = (const uint4*)(B + (size_t)(nBase + 16 + r) * D);

  v8f c00 = {}, c01 = {}, c10 = {}, c11 = {};
  const int kTiles = D >> 5;  // K=32 per WMMA
  for (int kt = 0; kt < kTiles; ++kt) {
    Frag a0, a1, b0, b1;
    // A-frag (ISA 16-bit A 16x32): dwords {4*hi+0..3} and {8+4*hi+0..3} of the K-tile
    a0.u4[0] = A0[kt*4 + hi];      a0.u4[1] = A0[kt*4 + 2 + hi];
    a1.u4[0] = A1[kt*4 + hi];      a1.u4[1] = A1[kt*4 + 2 + hi];
    // B-frag (K x N): lane n=r holds K = kt*32 + 8*hi .. +7 and kt*32 + 16 + 8*hi .. +7 of key row n
    b0.u4[0] = B0[kt*4 + hi];      b0.u4[1] = B0[kt*4 + 2 + hi];
    b1.u4[0] = B1[kt*4 + hi];      b1.u4[1] = B1[kt*4 + 2 + hi];
    if (kt + 1 < kTiles) {
      __builtin_prefetch(&A0[kt*4 + 4], 0, 1);
      __builtin_prefetch(&A1[kt*4 + 4], 0, 1);
    }
    c00 = __builtin_amdgcn_wmma_f32_16x16x32_f16(false, a0.h, false, b0.h, (short)0, c00, false, false);
    c01 = __builtin_amdgcn_wmma_f32_16x16x32_f16(false, a0.h, false, b1.h, (short)0, c01, false, false);
    c10 = __builtin_amdgcn_wmma_f32_16x16x32_f16(false, a1.h, false, b0.h, (short)0, c10, false, false);
    c11 = __builtin_amdgcn_wmma_f32_16x16x32_f16(false, a1.h, false, b1.h, (short)0, c11, false, false);
  }
#pragma unroll
  for (int j = 0; j < 8; ++j) {  // C/D: VGPR j -> rows j (lanes 0-15) / j+8 (lanes 16-31)
    int m0 = mBase + j + 8*hi;
    int m1 = m0 + 16;
    score[(size_t)m0*N + nBase + r]      = c00[j] * invKT;
    score[(size_t)m0*N + nBase + 16 + r] = c01[j] * invKT;
    score[(size_t)m1*N + nBase + r]      = c10[j] * invKT;
    score[(size_t)m1*N + nBase + 16 + r] = c11[j] * invKT;
  }
}

// ---------------- per-row softmax + argmax + key gather (N == blockDim == 512) ----------------
__global__ void k_softmax_argmax(const float* __restrict__ score, float* __restrict__ P,
                                 float* __restrict__ indices_f, float* __restrict__ emb,
                                 const float* __restrict__ keys_n, float* __restrict__ rowmax,
                                 int N, int D) {
  __shared__ float sv[512];
  __shared__ int   si[512];
  int m = blockIdx.x, t = threadIdx.x;
  float v = score[(size_t)m * N + t];
  sv[t] = v; si[t] = t;
  __syncthreads();
  for (int s = 256; s > 0; s >>= 1) {
    if (t < s) {
      float o = sv[t + s]; int oi = si[t + s];
      if (o > sv[t] || (o == sv[t] && oi < si[t])) { sv[t] = o; si[t] = oi; }
    }
    __syncthreads();
  }
  float mx = sv[0]; int amax = si[0];
  __syncthreads();
  float e = __expf(v - mx);
  sv[t] = e;
  __syncthreads();
  for (int s = 256; s > 0; s >>= 1) { if (t < s) sv[t] += sv[t + s]; __syncthreads(); }
  float sum = sv[0];
  P[(size_t)m * N + t] = e / sum;
  if (t == 0) { indices_f[m] = (float)amax; rowmax[m] = mx; }
  if (t < D) emb[(size_t)m * D + t] = keys_n[(size_t)amax * D + t];
}

__global__ void k_fill(float* __restrict__ p, int n, float v) {
  int i = blockIdx.x * blockDim.x + threadIdx.x;
  if (i < n) p[i] = v;
}

// u0[n] += sum_m exp((score[m][n]-rowmax[m])/eps)   (b==1; N==512, block=256, 2 cols/thread)
__global__ void k_colsum0(const float* __restrict__ score, const float* __restrict__ rowmax,
                          float* __restrict__ u, int M, int N, int rowsPerBlock) {
  int t = threadIdx.x;
  float acc0 = 0.f, acc1 = 0.f;
  int r0 = blockIdx.x * rowsPerBlock;
  int r1 = min(r0 + rowsPerBlock, M);
  for (int m = r0; m < r1; ++m) {
    const float* row = score + (size_t)m * N;
    float g = rowmax[m];
    acc0 += __expf((row[t]       - g) * INV_EPS);
    acc1 += __expf((row[t + 256] - g) * INV_EPS);
  }
  atomicAdd(&u[t], acc0);
  atomicAdd(&u[t + 256], acc1);
}

__global__ void k_a_update(float* __restrict__ a, const float* __restrict__ u, int n, float num) {
  int i = blockIdx.x * blockDim.x + threadIdx.x;
  if (i < n) a[i] = num / u[i];
}

// Fused Sinkhorn step, one wave per row (N == 512 == 32 lanes * 16):
//   z = sum_n a[n]*E'[m,n];  Z[m] = z;  b_m = (1/M)/z
//   if accumulate: u_next[n] += b_m*E'[m,n]  (register partials, one atomicAdd burst at end)
__global__ void k_sk_iter(const float* __restrict__ score, const float* __restrict__ a,
                          const float* __restrict__ rowmax, float* __restrict__ Z,
                          float* __restrict__ u_next, int M, int N,
                          int rowsPerBlock, int accumulate) {
  int wave = threadIdx.x >> 5, lane = threadIdx.x & 31;
  int wavesPerBlock = blockDim.x >> 5;
  float av[16], acc[16];
#pragma unroll
  for (int i = 0; i < 16; ++i) { av[i] = a[lane + 32 * i]; acc[i] = 0.f; }
  int r0 = blockIdx.x * rowsPerBlock;
  int r1 = min(r0 + rowsPerBlock, M);
  for (int m = r0 + wave; m < r1; m += wavesPerBlock) {
    const float* row = score + (size_t)m * N;
    float g = rowmax[m];
    float e[16];
    float z = 0.f;
#pragma unroll
    for (int i = 0; i < 16; ++i) {
      e[i] = __expf((row[lane + 32 * i] - g) * INV_EPS);
      z += av[i] * e[i];
    }
    z = wave_sum(z);
    float bm = (1.0f / (float)M) / z;
    if (lane == 0) Z[m] = z;
    if (accumulate) {
#pragma unroll
      for (int i = 0; i < 16; ++i) acc[i] += bm * e[i];
    }
  }
  if (accumulate) {
#pragma unroll
    for (int i = 0; i < 16; ++i) atomicAdd(&u_next[lane + 32 * i], acc[i]);
  }
}

// in-place: Q[m][n] = a[n]*exp((score-rowmax[m])/eps)/Z[m]  (score lives in Q output region)
__global__ void k_final_q(float* __restrict__ q, const float* __restrict__ a,
                          const float* __restrict__ Z, const float* __restrict__ rowmax,
                          int M, int N) {
  size_t id = (size_t)blockIdx.x * blockDim.x + threadIdx.x;
  if (id >= (size_t)M * N) return;
  int n = (int)(id % (size_t)N);
  size_t m = id / (size_t)N;
  float val = a[n] * __expf((q[id] - rowmax[m]) * INV_EPS) / Z[m];
  if (isnan(val)) val = 0.f;
  else if (isinf(val)) val = val > 0.f ? 1e8f : -1e8f;
  q[id] = val;
}

extern "C" void kernel_launch(void* const* d_in, const int* in_sizes, int n_in,
                              void* d_out, int out_size, void* d_ws, size_t ws_size,
                              hipStream_t stream) {
  const float* x    = (const float*)d_in[0];  // [B*T, D] flattened
  const float* keys = (const float*)d_in[1];  // [N, D]
  const int D = 256;
  const int M = in_sizes[0] / D;              // 32768
  const int N = in_sizes[1] / D;              // 512

  float* out      = (float*)d_out;
  float* out_idx  = out;                              // [M]
  float* out_emb  = out_idx + M;                      // [M*D]
  float* out_P    = out_emb + (size_t)M * D;          // [M*N]
  float* out_Q    = out_P   + (size_t)M * N;          // [M*N] -- holds score until k_final_q

  // workspace carve (~17.1 MB total)
  char* w = (char*)d_ws;
  auto carve = [&](size_t bytes) { char* p = w; w += (bytes + 255) & ~(size_t)255; return p; };
  float*    keys_n = (float*)   carve((size_t)N * D * 4);
  _Float16* keys_h = (_Float16*)carve((size_t)N * D * 2);
  _Float16* x_h    = (_Float16*)carve((size_t)M * D * 2);
  float*    rowmax = (float*)   carve((size_t)M * 4);
  float*    u      = (float*)   carve((size_t)N * 4);
  float*    a      = (float*)   carve((size_t)N * 4);
  float*    Z      = (float*)   carve((size_t)M * 4);

  const float invKT = (float)(log((double)(N - 1) * 0.9999 / 0.0001) / 2.0);

  // 1. normalize keys (fp32 + f16) and x (f16 only)
  k_l2norm<<<dim3((N + 7) / 8), dim3(256), 0, stream>>>(keys, keys_n, keys_h, N, D);
  k_l2norm<<<dim3((M + 7) / 8), dim3(256), 0, stream>>>(x, nullptr, x_h, M, D);

  // 2. score GEMM (WMMA), written into Q output region
  k_score_wmma<<<dim3(N / 32, M / 256), dim3(256), 0, stream>>>(x_h, keys_h, out_Q, M, N, D, invKT);

  // 3. softmax / argmax / gather (+ per-row max, reused for Sinkhorn stabilization)
  k_softmax_argmax<<<dim3(M), dim3(512), 0, stream>>>(out_Q, out_P, out_idx, out_emb, keys_n, rowmax, N, D);

  // 4. Sinkhorn (3 iterations), rank-1 scale vectors; row-update fused with next column-sum
  const int blocks = 128;
  const int rowsPerBlock = (M + blocks - 1) / blocks;
  const dim3 gN((N + 255) / 256), b256(256);

  k_fill<<<gN, b256, 0, stream>>>(u, N, 0.0f);
  k_colsum0<<<dim3(blocks), b256, 0, stream>>>(out_Q, rowmax, u, M, N, rowsPerBlock);      // u0 (b=1)
  k_a_update<<<gN, b256, 0, stream>>>(a, u, N, 1.0f / (float)N);                           // a0

  k_fill<<<gN, b256, 0, stream>>>(u, N, 0.0f);
  k_sk_iter<<<dim3(blocks), b256, 0, stream>>>(out_Q, a, rowmax, Z, u, M, N, rowsPerBlock, 1); // z0,b0 -> u1
  k_a_update<<<gN, b256, 0, stream>>>(a, u, N, 1.0f / (float)N);                           // a1

  k_fill<<<gN, b256, 0, stream>>>(u, N, 0.0f);
  k_sk_iter<<<dim3(blocks), b256, 0, stream>>>(out_Q, a, rowmax, Z, u, M, N, rowsPerBlock, 1); // z1,b1 -> u2
  k_a_update<<<gN, b256, 0, stream>>>(a, u, N, 1.0f / (float)N);                           // a2

  k_sk_iter<<<dim3(blocks), b256, 0, stream>>>(out_Q, a, rowmax, Z, u, M, N, rowsPerBlock, 0); // z2 only

  // 5. final Q (in-place over score)
  size_t total = (size_t)M * N;
  k_final_q<<<dim3((unsigned)((total + 255) / 256)), b256, 0, stream>>>(out_Q, a, Z, rowmax, M, N);
}
